// Transformer_3951369912878
// MI455X (gfx1250) — compile-verified, matrix operand order checked by reading, not run on hardware
//
#include <hip/hip_runtime.h>

// ---------------------------------------------------------------------------
// Types for CDNA5 WMMA (wave32): A/B = 16 bf16 per lane, C/D = 8 f32 per lane
// ---------------------------------------------------------------------------
typedef __bf16         bf16x16 __attribute__((ext_vector_type(16)));
typedef float          f32x8   __attribute__((ext_vector_type(8)));
typedef float          f32x4   __attribute__((ext_vector_type(4)));
typedef unsigned int   u32x4   __attribute__((ext_vector_type(4)));
typedef unsigned short u16x4   __attribute__((ext_vector_type(4)));

union FragCast {
    bf16x16 v;
    u32x4   q[2];
};

__device__ __forceinline__ unsigned short f2bf(float f) {
    // round-to-nearest-even fp32 -> bf16 bits
    unsigned int u = __float_as_uint(f);
    u += 0x7FFFu + ((u >> 16) & 1u);
    return (unsigned short)(u >> 16);
}

__device__ __forceinline__ u16x4 f2bf4(f32x4 v) {
    u16x4 h;
    h.x = f2bf(v.x); h.y = f2bf(v.y); h.z = f2bf(v.z); h.w = f2bf(v.w);
    return h;
}

// Load one bf16 WMMA operand fragment from an LDS tile stored row-major
// [rows][32 k] (16B-aligned rows).
// Per ISA: lanes 0-15 hold K=0..7 and 16..23; lanes 16-31 hold K=8..15 and 24..31.
__device__ __forceinline__ bf16x16 load_frag(const unsigned short* p0) {
    FragCast u;
    u.q[0] = *(const u32x4*)(p0);       // K base .. base+7
    u.q[1] = *(const u32x4*)(p0 + 16);  // K base+16 .. base+23
    return u.v;
}

// ---------------------------------------------------------------------------
// Generic batched GEMM: C = act( scale * (A @ B) + bias )
//   A: fp32 [M,K], row stride lda, batch offset = (z/Hdiv)*oAb + (z%Hdiv)*oAh
//   B: fp32; if bT==0 element (k,n) at Bm[k*ldb+n]; if bT==1 at Bm[n*ldb+k]
//   Block = 128 threads (4 waves); block tile 64x64; wave tile 32x32 (2x2 WMMA)
//   Double-buffered LDS staging (fp32 -> bf16), one barrier per K-step.
//   Staging is two-phase: all 8 global b128 loads issued first, then convert+store.
// ---------------------------------------------------------------------------
__global__ __launch_bounds__(128)
void gemm_wmma(const float* __restrict__ A, long oAb, long oAh, int lda,
               const float* __restrict__ Bm, long oBb, long oBh, int ldb, int bT,
               const float* __restrict__ bias,
               float* __restrict__ C, long oCb, long oCh, int ldc,
               int K, float scale, int act, int Hdiv)
{
    __shared__ alignas(16) unsigned short sA[2][64 * 32]; // [m][k]
    __shared__ alignas(16) unsigned short sB[2][64 * 32]; // [n][k]

    const int z  = blockIdx.z;
    const int bi = z / Hdiv;
    const int hi = z % Hdiv;
    A  += (size_t)bi * oAb + (size_t)hi * oAh;
    Bm += (size_t)bi * oBb + (size_t)hi * oBh;
    C  += (size_t)bi * oCb + (size_t)hi * oCh;

    const int tid  = threadIdx.x;
    const int lane = tid & 31;
    const int wave = tid >> 5;
    const int m0 = blockIdx.y * 64;
    const int n0 = blockIdx.x * 64;
    const int wm = (wave >> 1) * 32;  // wave row offset within block tile
    const int wn = (wave & 1) * 32;   // wave col offset within block tile

    f32x8 acc00 = {}, acc01 = {}, acc10 = {}, acc11 = {};

    const int lrow  = lane & 15;
    const int khalf = (lane >> 4) << 3; // 0 or 8

    // Stage one 64x32 A tile and one 64x32 (as [n][k]) B tile into buffer p.
    // Phase 1: issue all 8 global b128 loads. Phase 2: convert + LDS store.
    auto stage = [&](int k0, int p) {
        f32x4 va[4], vb[4];
        #pragma unroll
        for (int it = 0; it < 4; ++it) {
            int idx = (tid + it * 128) * 4;       // 0..2047, step 4
            int r = idx >> 5, c = idx & 31;       // A tile [64][32]
            va[it] = *(const f32x4*)(A + (size_t)(m0 + r) * lda + (k0 + c));
        }
        if (bT) {
            // B source is [N,K]: rows are k-contiguous, same pattern as A.
            #pragma unroll
            for (int it = 0; it < 4; ++it) {
                int idx = (tid + it * 128) * 4;
                int n = idx >> 5, c = idx & 31;
                vb[it] = *(const f32x4*)(Bm + (size_t)(n0 + n) * ldb + (k0 + c));
            }
            #pragma unroll
            for (int it = 0; it < 4; ++it) {
                int idx = (tid + it * 128) * 4;
                int r = idx >> 5, c = idx & 31;   // same index math for A and B
                *(u16x4*)(&sA[p][r * 32 + c]) = f2bf4(va[it]);
                *(u16x4*)(&sB[p][r * 32 + c]) = f2bf4(vb[it]);
            }
        } else {
            // B source is [K,N]: load 4 consecutive n, scatter into [n][k].
            #pragma unroll
            for (int it = 0; it < 4; ++it) {
                int idx = (tid + it * 128) * 4;
                int kk = idx >> 6, n = idx & 63;  // view as [32 k][64 n]
                vb[it] = *(const f32x4*)(Bm + (size_t)(k0 + kk) * ldb + (n0 + n));
            }
            #pragma unroll
            for (int it = 0; it < 4; ++it) {
                int idx = (tid + it * 128) * 4;
                int r = idx >> 5, c = idx & 31;
                *(u16x4*)(&sA[p][r * 32 + c]) = f2bf4(va[it]);
            }
            #pragma unroll
            for (int it = 0; it < 4; ++it) {
                int idx = (tid + it * 128) * 4;
                int kk = idx >> 6, n = idx & 63;
                sB[p][(n + 0) * 32 + kk] = f2bf(vb[it].x);
                sB[p][(n + 1) * 32 + kk] = f2bf(vb[it].y);
                sB[p][(n + 2) * 32 + kk] = f2bf(vb[it].z);
                sB[p][(n + 3) * 32 + kk] = f2bf(vb[it].w);
            }
        }
    };

    auto compute = [&](int p) {
        bf16x16 a0 = load_frag(&sA[p][(wm + 0  + lrow) * 32 + khalf]);
        bf16x16 a1 = load_frag(&sA[p][(wm + 16 + lrow) * 32 + khalf]);
        bf16x16 b0 = load_frag(&sB[p][(wn + 0  + lrow) * 32 + khalf]);
        bf16x16 b1 = load_frag(&sB[p][(wn + 16 + lrow) * 32 + khalf]);
        acc00 = __builtin_amdgcn_wmma_f32_16x16x32_bf16(false, a0, false, b0, (short)0, acc00, false, false);
        acc01 = __builtin_amdgcn_wmma_f32_16x16x32_bf16(false, a0, false, b1, (short)0, acc01, false, false);
        acc10 = __builtin_amdgcn_wmma_f32_16x16x32_bf16(false, a1, false, b0, (short)0, acc10, false, false);
        acc11 = __builtin_amdgcn_wmma_f32_16x16x32_bf16(false, a1, false, b1, (short)0, acc11, false, false);
    };

    // Double-buffered pipeline: one barrier per K-step.
    stage(0, 0);
    int p = 0;
    for (int k0 = 32; k0 < K; k0 += 32) {
        __syncthreads();        // buf[p] staged; buf[p^1] free (prev compute done)
        stage(k0, p ^ 1);       // issue next tile's global loads early
        compute(p);
        p ^= 1;
    }
    __syncthreads();
    compute(p);

    // Store: C/D layout — VGPR r: lanes 0-15 -> M=r, lanes 16-31 -> M=8+r; N = lane&15
    const int cn = lane & 15;
    const int rh = (lane >> 4) << 3; // 0 or 8
    auto store_tile = [&](f32x8 acc, int mt, int nt) {
        int gn = n0 + wn + nt + cn;
        float bval = bias ? bias[gn] : 0.0f;
        #pragma unroll
        for (int r = 0; r < 8; ++r) {
            float v = acc[r] * scale + bval;
            if (act) v = fmaxf(v, 0.0f);
            C[(size_t)(m0 + wm + mt + rh + r) * ldc + gn] = v;
        }
    };
    store_tile(acc00, 0,  0);
    store_tile(acc01, 0,  16);
    store_tile(acc10, 16, 0);
    store_tile(acc11, 16, 16);
}

// ---------------------------------------------------------------------------
// In-place masked softmax over score rows.
//   scores: [B*H, Sq, Sk]; mode 0: src key-pad mask [B,Sk]; mode 1: causal [B,Sq,Sk]
// ---------------------------------------------------------------------------
__global__ __launch_bounds__(128)
void softmax_mask(float* __restrict__ scores, const int* __restrict__ mask,
                  int mode, int Sq, int Sk, int Hdiv)
{
    __shared__ float red[128];
    const int q = blockIdx.x;
    const int z = blockIdx.y;
    const int b = z / Hdiv;
    float* row = scores + ((size_t)z * Sq + q) * Sk;
    const int* mrow = (mode == 0) ? (mask + (size_t)b * Sk)
                                  : (mask + ((size_t)b * Sq + q) * Sk);
    const int tid = threadIdx.x;

    float v[8];
    int cnt = 0;
    float lmax = -3.4e38f;
    for (int k = tid; k < Sk; k += 128) {
        float s = row[k];
        if (mrow[k] == 0) s = -1e9f;
        v[cnt++] = s;
        lmax = fmaxf(lmax, s);
    }
    red[tid] = lmax; __syncthreads();
    for (int off = 64; off > 0; off >>= 1) {
        if (tid < off) red[tid] = fmaxf(red[tid], red[tid + off]);
        __syncthreads();
    }
    float m = red[0]; __syncthreads();

    float lsum = 0.0f;
    cnt = 0;
    for (int k = tid; k < Sk; k += 128) {
        float e = __expf(v[cnt] - m);
        v[cnt++] = e;
        lsum += e;
    }
    red[tid] = lsum; __syncthreads();
    for (int off = 64; off > 0; off >>= 1) {
        if (tid < off) red[tid] += red[tid + off];
        __syncthreads();
    }
    float inv = 1.0f / red[0];
    cnt = 0;
    for (int k = tid; k < Sk; k += 128) row[k] = v[cnt++] * inv;
}

// ---------------------------------------------------------------------------
// Fused residual add + LayerNorm over last dim (D=1024). One block per row.
// ---------------------------------------------------------------------------
__global__ __launch_bounds__(256)
void layernorm(const float* __restrict__ x, const float* __restrict__ add,
               const float* __restrict__ g, const float* __restrict__ bta,
               float* __restrict__ out, int Dn)
{
    __shared__ float red[256];
    const int row = blockIdx.x;
    const int tid = threadIdx.x;
    const float* xr = x + (size_t)row * Dn;
    const float* ar = add ? add + (size_t)row * Dn : nullptr;

    float v[8];
    int cnt = 0;
    float lsum = 0.0f;
    for (int i = tid; i < Dn; i += 256) {
        float t = xr[i] + (ar ? ar[i] : 0.0f);
        v[cnt++] = t;
        lsum += t;
    }
    red[tid] = lsum; __syncthreads();
    for (int off = 128; off > 0; off >>= 1) {
        if (tid < off) red[tid] += red[tid + off];
        __syncthreads();
    }
    float mean = red[0] / (float)Dn; __syncthreads();

    float lvar = 0.0f;
    cnt = 0;
    for (int i = tid; i < Dn; i += 256) { float d = v[cnt++] - mean; lvar += d * d; }
    red[tid] = lvar; __syncthreads();
    for (int off = 128; off > 0; off >>= 1) {
        if (tid < off) red[tid] += red[tid + off];
        __syncthreads();
    }
    float rstd = rsqrtf(red[0] / (float)Dn + 1e-5f); __syncthreads();

    cnt = 0;
    for (int i = tid; i < Dn; i += 256)
        out[(size_t)row * Dn + i] = g[i] * (v[cnt++] - mean) * rstd + bta[i];
}

// ---------------------------------------------------------------------------
// Host-side orchestration
// ---------------------------------------------------------------------------
extern "C" void kernel_launch(void* const* d_in, const int* in_sizes, int n_in,
                              void* d_out, int out_size, void* d_ws, size_t ws_size,
                              hipStream_t stream)
{
    (void)in_sizes; (void)n_in; (void)out_size; (void)ws_size;

    const float* src         = (const float*)d_in[0];
    const float* tgt         = (const float*)d_in[1];
    const float* enc_qkvo_W  = (const float*)d_in[2];
    const float* enc_qkvo_b  = (const float*)d_in[3];
    const float* enc_ff_W1   = (const float*)d_in[4];
    const float* enc_ff_b1   = (const float*)d_in[5];
    const float* enc_ff_W2   = (const float*)d_in[6];
    const float* enc_ff_b2   = (const float*)d_in[7];
    const float* enc_ln_g    = (const float*)d_in[8];
    const float* enc_ln_b    = (const float*)d_in[9];
    const float* enc_final_g = (const float*)d_in[10];
    const float* enc_final_b = (const float*)d_in[11];
    const float* dec_self_W  = (const float*)d_in[12];
    const float* dec_self_b  = (const float*)d_in[13];
    const float* dec_cross_W = (const float*)d_in[14];
    const float* dec_cross_b = (const float*)d_in[15];
    const float* dec_ff_W1   = (const float*)d_in[16];
    const float* dec_ff_b1   = (const float*)d_in[17];
    const float* dec_ff_W2   = (const float*)d_in[18];
    const float* dec_ff_b2   = (const float*)d_in[19];
    const float* dec_ln_g    = (const float*)d_in[20];
    const float* dec_ln_b    = (const float*)d_in[21];
    const float* dec_final_g = (const float*)d_in[22];
    const float* dec_final_b = (const float*)d_in[23];
    const int*   src_mask    = (const int*)d_in[24];
    const int*   tgt_mask    = (const int*)d_in[25];

    float* outF = (float*)d_out;
    float* ws   = (float*)d_ws;

    constexpr int Lc = 4, Dc = 1024, Hc = 16, DFFc = 4096, Bc = 2, Sc = 512, DKc = 64;
    constexpr int TOK = Bc * Sc;                         // 1024 tokens
    constexpr size_t ND   = (size_t)TOK * Dc;            // 1,048,576
    constexpr size_t ATT1 = (size_t)Bc * Hc * Sc * Sc;   // per-layer attn elems
    constexpr size_t ATTL = ATT1 * Lc;

    // Workspace layout (floats)
    float* X    = ws;
    float* Qb   = X    + ND;
    float* Kbuf = Qb   + ND;
    float* Vbuf = Kbuf + ND;
    float* CTX  = Vbuf + ND;
    float* TMP  = CTX  + ND;
    float* HID  = TMP  + ND;                 // TOK x DFF
    float* ENC  = HID  + (size_t)TOK * DFFc;

    // Output layout (floats): dec_out | enc_attns | self_attns | cross_attns
    float* encAtt   = outF + ND;
    float* selfAtt  = encAtt + ATTL;
    float* crossAtt = selfAtt + ATTL;

    auto gemm = [&](const float* A, long oAb, long oAh, int lda,
                    const float* Bm, long oBb, long oBh, int ldb, int bT,
                    const float* bias, float* C, long oCb, long oCh, int ldc,
                    int M, int N, int K, float scale, int act, int batches, int Hdiv) {
        dim3 g(N / 64, M / 64, batches);
        gemm_wmma<<<g, 128, 0, stream>>>(A, oAb, oAh, lda, Bm, oBb, oBh, ldb, bT,
                                         bias, C, oCb, oCh, ldc, K, scale, act, Hdiv);
    };

    auto ln = [&](const float* x, const float* add, const float* g, const float* b, float* out) {
        layernorm<<<TOK, 256, 0, stream>>>(x, add, g, b, out, Dc);
    };

    // Multi-head attention: Q from Xq, K/V from Xkv; attn probs written to attOut slice.
    auto attention = [&](const float* Xq, const float* Xkv,
                         const float* W, const float* bvec,
                         float* attOut, const int* mask, int mode) {
        // Projections (per token, M = B*S)
        gemm(Xq,  0, 0, Dc, W + 0 * (size_t)Dc * Dc, 0, 0, Dc, 0, bvec + 0 * Dc, Qb,   0, 0, Dc, TOK, Dc, Dc, 1.f, 0, 1, 1);
        gemm(Xkv, 0, 0, Dc, W + 1 * (size_t)Dc * Dc, 0, 0, Dc, 0, bvec + 1 * Dc, Kbuf, 0, 0, Dc, TOK, Dc, Dc, 1.f, 0, 1, 1);
        gemm(Xkv, 0, 0, Dc, W + 2 * (size_t)Dc * Dc, 0, 0, Dc, 0, bvec + 2 * Dc, Vbuf, 0, 0, Dc, TOK, Dc, Dc, 1.f, 0, 1, 1);
        // scores = (Q @ K^T) / sqrt(DK), batched over B*H, written into d_out slice
        gemm(Qb,   (long)Sc * Dc, DKc, Dc,
             Kbuf, (long)Sc * Dc, DKc, Dc, /*bT=*/1, nullptr,
             attOut, (long)Hc * Sc * Sc, (long)Sc * Sc, Sc,
             Sc, Sc, DKc, 0.125f, 0, Bc * Hc, Hc);
        // mask + softmax in place
        softmax_mask<<<dim3(Sc, Bc * Hc), 128, 0, stream>>>(attOut, mask, mode, Sc, Sc, Hc);
        // ctx = attn @ V  -> [B,S,H*DK]
        gemm(attOut, (long)Hc * Sc * Sc, (long)Sc * Sc, Sc,
             Vbuf,   (long)Sc * Dc, DKc, Dc, /*bT=*/0, nullptr,
             CTX,    (long)Sc * Dc, DKc, Dc,
             Sc, DKc, Sc, 1.f, 0, Bc * Hc, Hc);
        // output projection
        gemm(CTX, 0, 0, Dc, W + 3 * (size_t)Dc * Dc, 0, 0, Dc, 0, bvec + 3 * Dc, TMP, 0, 0, Dc, TOK, Dc, Dc, 1.f, 0, 1, 1);
    };

    auto ffn = [&](const float* Xin, const float* W1, const float* b1,
                   const float* W2, const float* b2) {
        gemm(Xin, 0, 0, Dc,   W1, 0, 0, DFFc, 0, b1, HID, 0, 0, DFFc, TOK, DFFc, Dc,   1.f, 1 /*relu*/, 1, 1);
        gemm(HID, 0, 0, DFFc, W2, 0, 0, Dc,   0, b2, TMP, 0, 0, Dc,   TOK, Dc,   DFFc, 1.f, 0,          1, 1);
    };

    // ----------------- Encoder -----------------
    hipMemcpyAsync(X, src, ND * sizeof(float), hipMemcpyDeviceToDevice, stream);
    for (int l = 0; l < Lc; ++l) {
        attention(X, X, enc_qkvo_W + (size_t)l * 4 * Dc * Dc, enc_qkvo_b + (size_t)l * 4 * Dc,
                  encAtt + (size_t)l * ATT1, src_mask, 0);
        ln(X, TMP, enc_ln_g + (size_t)l * 2 * Dc,      enc_ln_b + (size_t)l * 2 * Dc,      X);
        ffn(X, enc_ff_W1 + (size_t)l * Dc * DFFc, enc_ff_b1 + (size_t)l * DFFc,
               enc_ff_W2 + (size_t)l * DFFc * Dc, enc_ff_b2 + (size_t)l * Dc);
        ln(X, TMP, enc_ln_g + (size_t)l * 2 * Dc + Dc, enc_ln_b + (size_t)l * 2 * Dc + Dc, X);
    }
    ln(X, nullptr, enc_final_g, enc_final_b, ENC);

    // ----------------- Decoder -----------------
    hipMemcpyAsync(X, tgt, ND * sizeof(float), hipMemcpyDeviceToDevice, stream);
    for (int l = 0; l < Lc; ++l) {
        attention(X, X, dec_self_W + (size_t)l * 4 * Dc * Dc, dec_self_b + (size_t)l * 4 * Dc,
                  selfAtt + (size_t)l * ATT1, tgt_mask, 1);
        ln(X, TMP, dec_ln_g + (size_t)l * 3 * Dc,          dec_ln_b + (size_t)l * 3 * Dc,          X);
        attention(X, ENC, dec_cross_W + (size_t)l * 4 * Dc * Dc, dec_cross_b + (size_t)l * 4 * Dc,
                  crossAtt + (size_t)l * ATT1, src_mask, 0);
        ln(X, TMP, dec_ln_g + (size_t)l * 3 * Dc + Dc,     dec_ln_b + (size_t)l * 3 * Dc + Dc,     X);
        ffn(X, dec_ff_W1 + (size_t)l * Dc * DFFc, dec_ff_b1 + (size_t)l * DFFc,
               dec_ff_W2 + (size_t)l * DFFc * Dc, dec_ff_b2 + (size_t)l * Dc);
        ln(X, TMP, dec_ln_g + (size_t)l * 3 * Dc + 2 * Dc, dec_ln_b + (size_t)l * 3 * Dc + 2 * Dc, X);
    }
    ln(X, nullptr, dec_final_g, dec_final_b, outF); // dec_out at offset 0
}
